// HybridConv_65481071409070
// MI455X (gfx1250) — compile-verified, matrix operand order checked by reading, not run on hardware
//
#include <hip/hip_runtime.h>

typedef __attribute__((ext_vector_type(2)))  float    v2f;
typedef __attribute__((ext_vector_type(8)))  float    v8f;
typedef __attribute__((ext_vector_type(16))) _Float16 v16h;

#define LOG2E 1.4426950408889634f

#if __has_builtin(__builtin_amdgcn_wmma_f32_16x16x4_f32)
#define USE_WMMA_F32X4 1
#else
#define USE_WMMA_F32X4 0
#endif

// ---- lane-permute helpers (ds_swizzle, group-of-32 xor mode) ----
template <int IMM>
__device__ __forceinline__ float swzf(float v) {
  return __int_as_float(__builtin_amdgcn_ds_swizzle(__float_as_int(v), IMM));
}
template <int IMM>
__device__ __forceinline__ int swzi(int v) {
  return __builtin_amdgcn_ds_swizzle(v, IMM);
}

// ---- fast transcendental wrappers (gfx1250 v_exp/v_rcp/v_tanh) ----
__device__ __forceinline__ float fexp2(float x) {
#if __has_builtin(__builtin_amdgcn_exp2f)
  return __builtin_amdgcn_exp2f(x);
#else
  return exp2f(x);
#endif
}
__device__ __forceinline__ float frcp(float x) {
#if __has_builtin(__builtin_amdgcn_rcpf)
  return __builtin_amdgcn_rcpf(x);
#else
  return 1.0f / x;
#endif
}
__device__ __forceinline__ float ftanh(float x) {
#if __has_builtin(__builtin_amdgcn_tanhf)
  return __builtin_amdgcn_tanhf(x);
#else
  float e = fexp2(x * (2.0f * LOG2E));         // exp(2x)
  return 1.0f - 2.0f * frcp(e + 1.0f);
#endif
}

// Combine two per-lane partial rows into one register while halving the
// lane-footprint of each: lanes with (lid&M)==0 keep a-rows, ==1 keep b-rows.
template <int M, int IMM>
__device__ __forceinline__ float combine2(float a, float b, int lid) {
  const bool s = (lid & M) != 0;
  const float lo = s ? b : a;
  const float hv = s ? a : b;
  return lo + swzf<IMM>(hv);
}

#if !USE_WMMA_F32X4
__device__ __forceinline__ unsigned pack_h2(float x, float y) {
  typedef _Float16 h2 __attribute__((ext_vector_type(2)));
  h2 p; p.x = (_Float16)x; p.y = (_Float16)y;
  unsigned u; __builtin_memcpy(&u, &p, 4);
  return u;
}
#endif

// One wave processes 16 patches per iteration:
//   feats[16x8] (RBF, algebraically reduced) -> WMMA x2 (K=4 f32) against
//   w1[8x16] with b1 preloaded into C -> tanh -> ds_swizzle tree-reduce
//   against w2 -> one coalesced 64B store of 16 outputs.
__global__ __launch_bounds__(256) void hybridconv_wmma_kernel(
    const float* __restrict__ data,   const float* __restrict__ conv_w,
    const float* __restrict__ conv_b, const float* __restrict__ basis,
    const float* __restrict__ w1,     const float* __restrict__ b1,
    const float* __restrict__ w2,     const float* __restrict__ b2,
    float* __restrict__ out, int n)
{
  const int lid  = (int)threadIdx.x & 31;
  const int land = lid & 15;          // patch within tile / hidden column
  const int hi   = lid >> 4;          // which K-half of the A operand
  const int kh   = hi << 1;           // basis-row base for this lane: {kh,kh+1,kh+4,kh+5}

  // conv weights (wave-uniform)
  const float cw0 = conv_w[0], cw1 = conv_w[1], cw2 = conv_w[2], cw3 = conv_w[3];
  const float cb  = conv_b[0];

  // Per-lane RBF constants: feats[b] = exp2(act*tse[b] + (nse[b] - 4*act^2*log2e))
  float tse[4], nse[4];
#pragma unroll
  for (int j = 0; j < 4; ++j) {
    const int row = kh + (j & 1) + ((j >> 1) << 2);   // {kh, kh+1, kh+4, kh+5}
    const float e0 = basis[row * 4 + 0];
    const float e1 = basis[row * 4 + 1];
    const float e2 = basis[row * 4 + 2];
    const float e3 = basis[row * 4 + 3];
    const float S1 = e0 + e1 + e2 + e3;
    const float S2 = e0 * e0 + e1 * e1 + e2 * e2 + e3 * e3;
    tse[j] = 2.0f * S1 * LOG2E;
    nse[j] = -S2 * LOG2E;
  }

#if USE_WMMA_F32X4
  // B operand (4x16 f32): VGPR j, lanes0-15 hold K=j row, lanes16-31 hold K=j+2
  v2f bA, bB;
  bA.x = w1[(kh + 0) * 16 + land];
  bA.y = w1[(kh + 1) * 16 + land];
  bB.x = w1[(kh + 4) * 16 + land];   // second WMMA covers basis rows 4..7
  bB.y = w1[(kh + 5) * 16 + land];
#else
  // f16 fallback B (32x16): lanes0-15 hold K=0..15 (pairs per VGPR); rows>=8 zero
  v16h B16;
#pragma unroll
  for (int j = 0; j < 8; ++j) {
    const bool use = (!hi) && (j < 4);
    const float lo = use ? w1[(2 * j + 0) * 16 + land] : 0.0f;
    const float hf = use ? w1[(2 * j + 1) * 16 + land] : 0.0f;
    B16[2 * j + 0] = (_Float16)lo;
    B16[2 * j + 1] = (_Float16)hf;
  }
#endif

  const float cinit = b1[land];              // bias folded into C
  const float wn    = w2[land];              // layer-2 weight for this lane's column
  const float b2s   = b2[0] * (1.0f / 16.0f);// distributed so 16-lane sum adds b2 once

  const int ntiles        = (n + 15) >> 4;
  const int wavesPerBlock = (int)blockDim.x >> 5;
  const int gwave         = (int)blockIdx.x * wavesPerBlock + ((int)threadIdx.x >> 5);
  const int nwaves        = (int)gridDim.x * wavesPerBlock;

  const float4* __restrict__ d4 = reinterpret_cast<const float4*>(data);

  for (int tile = gwave; tile < ntiles; tile += nwaves) {
    const int p  = (tile << 4) + land;       // lanes m and m+16 share patch m (cheap dup)
    const int pc = (p < n) ? p : (n - 1);
    const float4 d = d4[pc];

    // speculative prefetch of this wave's next tile (global_prefetch_b8)
    int pt = tile + nwaves;
    pt = (pt < ntiles) ? pt : tile;
    __builtin_prefetch(reinterpret_cast<const char*>(d4 + ((pt << 4) + land)), 0, 0);

    // conv -> sigmoid
    const float logit = fmaf(cw0, d.x, fmaf(cw1, d.y, fmaf(cw2, d.z, fmaf(cw3, d.w, cb))));
    const float act   = frcp(1.0f + fexp2(-logit * LOG2E));
    const float qe    = -4.0f * act * act * LOG2E;

    // reduced RBF: 3 ops + exp2 per basis row
    const float f0 = fexp2(fmaf(act, tse[0], nse[0] + qe));
    const float f1 = fexp2(fmaf(act, tse[1], nse[1] + qe));
    const float f2 = fexp2(fmaf(act, tse[2], nse[2] + qe));
    const float f3 = fexp2(fmaf(act, tse[3], nse[3] + qe));

    v8f c;
#pragma unroll
    for (int r = 0; r < 8; ++r) c[r] = cinit;

#if USE_WMMA_F32X4
    // A (16x4 f32): lanes0-15 VGPR0/1 = K0/K1, lanes16-31 = K2/K3 -> already in place
    v2f a1; a1.x = f0; a1.y = f1;            // basis rows 0..3
    v2f a2; a2.x = f2; a2.y = f3;            // basis rows 4..7
    c = __builtin_amdgcn_wmma_f32_16x16x4_f32(false, a1, false, bA, (short)0, c, false, false);
    c = __builtin_amdgcn_wmma_f32_16x16x4_f32(false, a2, false, bB, (short)0, c, false, false);
#else
    // f16 fallback: assemble A (16x32, K=0..7 live) on lanes 0-15
    unsigned pa = pack_h2(f0, f1);           // rows {0,1} (lo lanes) / {2,3} (hi lanes)
    unsigned pb = pack_h2(f2, f3);           // rows {4,5} / {6,7}
    unsigned qa = (unsigned)swzi<0x401F>((int)pa);  // pull partner half (xor 16)
    unsigned qb = (unsigned)swzi<0x401F>((int)pb);
    const unsigned zm = hi ? 0u : 0xFFFFFFFFu;      // K=8..15 half must be zero
    pa &= zm; qa &= zm; pb &= zm; qb &= zm;
    unsigned au[8] = {pa, qa, pb, qb, 0u, 0u, 0u, 0u};
    v16h A16; __builtin_memcpy(&A16, au, 32);
    c = __builtin_amdgcn_wmma_f32_16x16x32_f16(false, A16, false, B16, (short)0, c, false, false);
#endif

    // tanh + w2 weighting (+ b2/16 so the 16-lane sum carries the bias once)
    const float s0 = fmaf(ftanh(c[0]), wn, b2s);
    const float s1 = fmaf(ftanh(c[1]), wn, b2s);
    const float s2 = fmaf(ftanh(c[2]), wn, b2s);
    const float s3 = fmaf(ftanh(c[3]), wn, b2s);
    const float s4 = fmaf(ftanh(c[4]), wn, b2s);
    const float s5 = fmaf(ftanh(c[5]), wn, b2s);
    const float s6 = fmaf(ftanh(c[6]), wn, b2s);
    const float s7 = fmaf(ftanh(c[7]), wn, b2s);

    // packed butterfly tree: 8 regs x 16 lanes -> 1 reg, row = lane&7 per half
    const float t0 = combine2<1, 0x041F>(s0, s1, lid);
    const float t1 = combine2<1, 0x041F>(s2, s3, lid);
    const float t2 = combine2<1, 0x041F>(s4, s5, lid);
    const float t3 = combine2<1, 0x041F>(s6, s7, lid);
    const float u0 = combine2<2, 0x081F>(t0, t1, lid);
    const float u1 = combine2<2, 0x081F>(t2, t3, lid);
    float v        = combine2<4, 0x101F>(u0, u1, lid);
    v += swzf<0x201F>(v);                    // fold the last duplicate pair

    // lanes 0-7 hold out[0..7], lanes 16-23 hold out[8..15] -> one 64B store
    const int o = (tile << 4) + (lid & 7) + (hi << 3);
    if (((lid & 8) == 0) && (o < n)) out[o] = v;
  }
}

extern "C" void kernel_launch(void* const* d_in, const int* in_sizes, int n_in,
                              void* d_out, int out_size, void* d_ws, size_t ws_size,
                              hipStream_t stream) {
  const float* data   = (const float*)d_in[0];
  const float* conv_w = (const float*)d_in[1];
  const float* conv_b = (const float*)d_in[2];
  const float* basis  = (const float*)d_in[3];
  const float* w1     = (const float*)d_in[4];
  const float* b1     = (const float*)d_in[5];
  const float* w2     = (const float*)d_in[6];
  const float* b2     = (const float*)d_in[7];
  float* out = (float*)d_out;

  const int n = in_sizes[0] / 4;             // N patches (2x2 floats each)
  const int ntiles = (n + 15) / 16;
  const int block = 256;                     // 8 wave32's per workgroup
  const int wavesPerBlock = block / 32;
  int blocks = (ntiles + wavesPerBlock - 1) / wavesPerBlock;
  if (blocks > 4096) blocks = 4096;          // persistent-ish; ~16 tiles per wave
  if (blocks < 1) blocks = 1;

  hipLaunchKernelGGL(hybridconv_wmma_kernel, dim3(blocks), dim3(block), 0, stream,
                     data, conv_w, conv_b, basis, w1, b1, w2, b2, out, n);
}
